// GraphEvoBlock_53145925320870
// MI455X (gfx1250) — compile-verified, matrix operand order checked by reading, not run on hardware
//
#include <hip/hip_runtime.h>
#include <hip/hip_bf16.h>
#include <math.h>

typedef __attribute__((ext_vector_type(16))) __bf16 v16bf;
typedef __attribute__((ext_vector_type(4)))  __bf16 v4bf;
typedef __attribute__((ext_vector_type(8)))  float  v8f;

#define LRELU_SLOPE 0.2f
#define BN_EPS 1e-5f

// ---------- helpers ----------
__device__ __forceinline__ unsigned f2key(float f) {
  unsigned u = __float_as_uint(f);
  return (u & 0x80000000u) ? ~u : (u | 0x80000000u);   // monotonic float->u32
}
__device__ __forceinline__ float key2f(unsigned k) {
  unsigned u = (k & 0x80000000u) ? (k ^ 0x80000000u) : ~k;
  return __uint_as_float(u);
}
__device__ __forceinline__ float waveSum(float v) {
#pragma unroll
  for (int off = 16; off > 0; off >>= 1) v += __shfl_xor(v, off, 32);
  return v;
}

// ---------- init (runs every call: deterministic) ----------
__global__ void k_init(float* hacc, float* zacc, float* denom, unsigned* maxkey,
                       float* deg, float* stats1, float* stats2, int N) {
  long long idx = (long long)blockIdx.x * blockDim.x + threadIdx.x;
  if (idx < (long long)N * 128) hacc[idx] = 0.f;
  if (idx < (long long)N * 32)  zacc[idx] = 0.f;
  if (idx < (long long)N * 4) { denom[idx] = 0.f; maxkey[idx] = 0x007FFFFFu; } // key(-inf)
  if (idx < N) deg[idx] = 0.f;
  if (idx < 256) stats1[idx] = 0.f;
  else if (idx < 320) stats2[idx - 256] = 0.f;
}

// ---------- bf16 WMMA GEMM: D[M,NC] = A[M,K] @ B[K,NC] (+bias) ----------
// block = 256 threads (8 waves). Block tile: 128 (M) x 32 (N); each wave owns a
// 16(M)x32(N) strip = two 16x16 WMMA tiles sharing one A fragment. LDS tiles
// are double-buffered and stored pre-permuted into the ISA's per-lane WMMA
// fragment layout so a fragment load is one contiguous 32-byte LDS read
// (2 x ds_load_b128). Fragment inverse map for element k (0..31): q=k>>3,
// half=q&1, j=(q>>1)*8+(k&7); lane = half*16 + (row|col within 16).
template <int K, int NC>
__global__ void k_wmma_gemm(const float* __restrict__ A, const float* __restrict__ B,
                            const float* __restrict__ bias, float* __restrict__ D, int M) {
  __shared__ __align__(16) __bf16 sA[2][8 * 512];  // 2 x 8 wave-tiles x (32 lanes x 16)
  __shared__ __align__(16) __bf16 sB[2][2 * 512];  // 2 x 2 col-tiles  x (32 lanes x 16)
  const int t = threadIdx.x;
  const int wave = t >> 5, lane = t & 31;
  const int half = lane >> 4, nl = lane & 15;
  const int mBlock = blockIdx.x * 128;
  const int nBase  = blockIdx.y * 32;
  const bool full = (mBlock + 128) <= M;

  // A staging: 4 float4 per thread; pointers + LDS offsets are loop-invariant.
  const float* Ap[4]; int offA[4]; bool zr[4];
#pragma unroll
  for (int i = 0; i < 4; ++i) {
    int idx = t + i * 256;                   // 1024 float4 slots = 128x32 f32
    int r = idx >> 3, c4 = (idx & 7) * 4;
    int gr = mBlock + r;
    zr[i] = (gr >= M);
    int grc = zr[i] ? (M - 1) : gr;          // clamped, unconditional load
    Ap[i] = A + (long long)grc * K + c4;
    int q = c4 >> 3, rem = c4 & 7;
    offA[i] = (r >> 4) * 512 + ((q & 1) * 16 + (r & 15)) * 16 + (q >> 1) * 8 + rem;
  }
  // B staging: lane-coalesced along n, wave picks k-quad; offsets become
  // compile-time immediates thanks to templated NC.
  const int bn = t & 31, bk = (t >> 5) * 4;
  const float* Bp = B + (long long)bk * NC + nBase + bn;
  int offB;
  {
    int q = bk >> 3, rem = bk & 7;
    offB = (bn >> 4) * 512 + ((q & 1) * 16 + (bn & 15)) * 16 + (q >> 1) * 8 + rem;
  }

  v8f acc0 = {}, acc1 = {};
#pragma unroll
  for (int kb = 0; kb < K; kb += 32) {
    const int buf = (kb >> 5) & 1;
    float4 av[4];
#pragma unroll
    for (int i = 0; i < 4; ++i) av[i] = *(const float4*)(Ap[i] + kb);
    float bv[4];
#pragma unroll
    for (int i = 0; i < 4; ++i) bv[i] = Bp[(long long)(kb + i) * NC];
    if (!full) {                             // uniform branch: tail block only
#pragma unroll
      for (int i = 0; i < 4; ++i)
        if (zr[i]) { av[i].x = 0.f; av[i].y = 0.f; av[i].z = 0.f; av[i].w = 0.f; }
    }
#pragma unroll
    for (int i = 0; i < 4; ++i) {
      v4bf pk;
      pk[0] = (__bf16)av[i].x; pk[1] = (__bf16)av[i].y;
      pk[2] = (__bf16)av[i].z; pk[3] = (__bf16)av[i].w;
      *(v4bf*)&sA[buf][offA[i]] = pk;        // one ds_store_b64
    }
    {
      v4bf pk;
      pk[0] = (__bf16)bv[0]; pk[1] = (__bf16)bv[1];
      pk[2] = (__bf16)bv[2]; pk[3] = (__bf16)bv[3];
      *(v4bf*)&sB[buf][offB] = pk;
    }
    __syncthreads();                         // single barrier per K-step (ping-pong)
    v16bf a  = *(const v16bf*)&sA[buf][wave * 512 + lane * 16];
    v16bf b0 = *(const v16bf*)&sB[buf][lane * 16];
    v16bf b1 = *(const v16bf*)&sB[buf][512 + lane * 16];
    acc0 = __builtin_amdgcn_wmma_f32_16x16x32_bf16(false, a, false, b0,
                                                   (short)0, acc0, false, false);
    acc1 = __builtin_amdgcn_wmma_f32_16x16x32_bf16(false, a, false, b1,
                                                   (short)0, acc1, false, false);
  }

#pragma unroll
  for (int r = 0; r < 8; ++r) {              // D layout: m = r + 8*half, n = nl
    int gm = mBlock + wave * 16 + (r + 8 * half);
    if (gm < M) {
      int gn0 = nBase + nl, gn1 = gn0 + 16;
      D[(long long)gm * NC + gn0] = acc0[r] + (bias ? bias[gn0] : 0.f);
      D[(long long)gm * NC + gn1] = acc1[r] + (bias ? bias[gn1] : 0.f);
    }
  }
}

// ---------- GATv2 attention passes (one edge per wave / per thread) ----------
__global__ void k_edge_logits(const int* __restrict__ e0, const int* __restrict__ e1,
                              const float* __restrict__ xl, const float* __restrict__ xr,
                              const float* __restrict__ att, float* __restrict__ logits,
                              unsigned* __restrict__ maxkey, float* __restrict__ deg,
                              int E, int N) {
  int eg = blockIdx.x * 8 + (threadIdx.x >> 5);
  int lane = threadIdx.x & 31;
  if (eg >= E + N) return;
  int s = (eg < E) ? e0[eg] : (eg - E);
  int d = (eg < E) ? e1[eg] : (eg - E);
  const float* pl = xl + (long long)s * 128;
  const float* pr = xr + (long long)d * 128;
  float lg[4];
#pragma unroll
  for (int h = 0; h < 4; ++h) {
    float v = pl[h * 32 + lane] + pr[h * 32 + lane];
    v = (v > 0.f) ? v : LRELU_SLOPE * v;
    lg[h] = waveSum(v * att[h * 32 + lane]);
  }
  if (lane == 0) {
#pragma unroll
    for (int h = 0; h < 4; ++h) {
      logits[(long long)eg * 4 + h] = lg[h];
      atomicMax(&maxkey[d * 4 + h], f2key(lg[h]));
    }
    atomicAdd(&deg[d], 1.f);   // in-degree incl. self loop (for GCN norm)
  }
}

__global__ void k_edge_p(const int* __restrict__ e1, const unsigned* __restrict__ maxkey,
                         float* __restrict__ pbuf, float* __restrict__ denom, int E, int N) {
  long long idx = (long long)blockIdx.x * blockDim.x + threadIdx.x;
  if (idx >= (long long)(E + N) * 4) return;
  int eg = (int)(idx >> 2), h = (int)(idx & 3);
  int d = (eg < E) ? e1[eg] : (eg - E);
  float p = __expf(pbuf[idx] - key2f(maxkey[d * 4 + h]));
  pbuf[idx] = p;
  atomicAdd(&denom[d * 4 + h], p);
}

__global__ void k_edge_scatter(const int* __restrict__ e0, const int* __restrict__ e1,
                               const float* __restrict__ xl, const float* __restrict__ pbuf,
                               const float* __restrict__ denom, float* __restrict__ hacc,
                               int E, int N) {
  int eg = blockIdx.x * 8 + (threadIdx.x >> 5);
  int lane = threadIdx.x & 31;
  if (eg >= E + N) return;
  int s = (eg < E) ? e0[eg] : (eg - E);
  int d = (eg < E) ? e1[eg] : (eg - E);
#pragma unroll
  for (int h = 0; h < 4; ++h) {
    float alpha = pbuf[(long long)eg * 4 + h] / (denom[d * 4 + h] + 1e-16f);
    atomicAdd(&hacc[(long long)d * 128 + h * 32 + lane],
              alpha * xl[(long long)s * 128 + h * 32 + lane]);
  }
}

// ---------- h = relu(acc + gat_bias) + x ; BN1 batch stats ----------
__global__ void k_h_stats(const float* __restrict__ x, const float* __restrict__ gat_bias,
                          float* __restrict__ hacc, float* __restrict__ stats1, int N) {
  int c = threadIdx.x;          // 128 channels
  int n0 = blockIdx.x * 32;
  float sum = 0.f, sq = 0.f;
  for (int i = 0; i < 32; ++i) {
    int n = n0 + i;
    if (n < N) {
      long long idx = (long long)n * 128 + c;
      float v = fmaxf(hacc[idx] + gat_bias[c], 0.f) + x[idx];
      hacc[idx] = v;
      sum += v; sq += v * v;
    }
  }
  atomicAdd(&stats1[c], sum);
  atomicAdd(&stats1[128 + c], sq);
}

__global__ void k_finalize(const float* __restrict__ stats, const float* __restrict__ gamma,
                           const float* __restrict__ beta, float* __restrict__ ss,
                           int C, int N) {
  int c = threadIdx.x;
  if (c >= C) return;
  float mean = stats[c] / (float)N;
  float var  = stats[C + c] / (float)N - mean * mean;
  float sc = gamma[c] * rsqrtf(var + BN_EPS);
  ss[c] = sc;
  ss[C + c] = beta[c] - mean * sc;
}

__global__ void k_affine(float* __restrict__ buf, const float* __restrict__ ss,
                         int C, long long tot) {
  long long idx = (long long)blockIdx.x * blockDim.x + threadIdx.x;
  if (idx >= tot) return;
  int c = (int)(idx & (C - 1));
  buf[idx] = buf[idx] * ss[c] + ss[C + c];
}

__global__ void k_dinv(float* deg, int N) {
  int i = blockIdx.x * blockDim.x + threadIdx.x;
  if (i >= N) return;
  float d = deg[i];
  deg[i] = (d > 0.f) ? rsqrtf(d) : 0.f;
}

// ---------- GCN scatter: zacc[dst] += dinv[s]*dinv[d] * hW[src] ----------
__global__ void k_gcn_scatter(const int* __restrict__ e0, const int* __restrict__ e1,
                              const float* __restrict__ dinv, const float* __restrict__ hW,
                              float* __restrict__ zacc, int E, int N) {
  int eg = blockIdx.x * 8 + (threadIdx.x >> 5);
  int lane = threadIdx.x & 31;
  if (eg >= E + N) return;
  int s = (eg < E) ? e0[eg] : (eg - E);
  int d = (eg < E) ? e1[eg] : (eg - E);
  float norm = dinv[s] * dinv[d];
  atomicAdd(&zacc[(long long)d * 32 + lane], norm * hW[(long long)s * 32 + lane]);
}

__global__ void k_z_stats(const float* __restrict__ bg, float* __restrict__ zacc,
                          float* __restrict__ stats2, int N) {
  int c = threadIdx.x;          // 32 channels
  int n0 = blockIdx.x * 64;
  float sum = 0.f, sq = 0.f;
  for (int i = 0; i < 64; ++i) {
    int n = n0 + i;
    if (n < N) {
      long long idx = (long long)n * 32 + c;
      float v = fmaxf(zacc[idx] + bg[c], 0.f);
      zacc[idx] = v;
      sum += v; sq += v * v;
    }
  }
  atomicAdd(&stats2[c], sum);
  atomicAdd(&stats2[32 + c], sq);
}

// ---------- GAE decode: sigmoid(<z[e0], z[e1]>) over original edges ----------
__global__ void k_decode(const int* __restrict__ e0, const int* __restrict__ e1,
                         const float* __restrict__ z, float* __restrict__ probs, int E) {
  int eg = blockIdx.x * 8 + (threadIdx.x >> 5);
  int lane = threadIdx.x & 31;
  if (eg >= E) return;
  int a = e0[eg], b = e1[eg];
  float s = waveSum(z[(long long)a * 32 + lane] * z[(long long)b * 32 + lane]);
  if (lane == 0) probs[eg] = 1.f / (1.f + __expf(-s));
}

// ---------- launch ----------
extern "C" void kernel_launch(void* const* d_in, const int* in_sizes, int n_in,
                              void* d_out, int out_size, void* d_ws, size_t ws_size,
                              hipStream_t stream) {
  const float* x        = (const float*)d_in[0];
  const int*   ei       = (const int*)d_in[1];
  const float* Wl       = (const float*)d_in[2];
  const float* bl       = (const float*)d_in[3];
  const float* Wr       = (const float*)d_in[4];
  const float* br       = (const float*)d_in[5];
  const float* att      = (const float*)d_in[6];
  const float* gat_bias = (const float*)d_in[7];
  const float* gamma1   = (const float*)d_in[8];
  const float* beta1    = (const float*)d_in[9];
  const float* Wg       = (const float*)d_in[10];
  const float* bg       = (const float*)d_in[11];
  const float* gamma2   = (const float*)d_in[12];
  const float* beta2    = (const float*)d_in[13];

  const int N  = in_sizes[0] / 128;
  const int E  = in_sizes[1] / 2;
  const int EP = E + N;                 // with self loops
  const int* e0 = ei;
  const int* e1 = ei + E;

  float* out_h     = (float*)d_out;               // [N,128]; doubles as GAT accumulator
  float* out_probs = out_h + (size_t)N * 128;     // [E]

  float* ws = (float*)d_ws;
  size_t off = 0;
  float*    xl     = ws + off; off += (size_t)N * 128;
  float*    xr     = ws + off; off += (size_t)N * 128;
  float*    pbuf   = ws + off; off += (size_t)EP * 4;   // logits, then softmax numerators
  unsigned* maxkey = (unsigned*)(ws + off); off += (size_t)N * 4;
  float*    denom  = ws + off; off += (size_t)N * 4;
  float*    deg    = ws + off; off += (size_t)N;        // degree -> dinv (in place)
  float*    hW     = ws + off; off += (size_t)N * 32;
  float*    zacc   = ws + off; off += (size_t)N * 32;   // z accumulator -> z (in place)
  float*    stats1 = ws + off; off += 256;
  float*    ss1    = ws + off; off += 256;
  float*    stats2 = ws + off; off += 64;
  float*    ss2    = ws + off; off += 64;
  (void)n_in; (void)out_size; (void)ws_size;

  {
    long long tot = (long long)N * 128;
    k_init<<<(int)((tot + 255) / 256), 256, 0, stream>>>(out_h, zacc, denom, maxkey,
                                                         deg, stats1, stats2, N);
  }
  {
    dim3 g((N + 127) / 128, 128 / 32);
    k_wmma_gemm<128, 128><<<g, 256, 0, stream>>>(x, Wl, bl, xl, N);
    k_wmma_gemm<128, 128><<<g, 256, 0, stream>>>(x, Wr, br, xr, N);
  }
  k_edge_logits<<<(EP + 7) / 8, 256, 0, stream>>>(e0, e1, xl, xr, att, pbuf, maxkey, deg, E, N);
  {
    long long tot = (long long)EP * 4;
    k_edge_p<<<(int)((tot + 255) / 256), 256, 0, stream>>>(e1, maxkey, pbuf, denom, E, N);
  }
  k_edge_scatter<<<(EP + 7) / 8, 256, 0, stream>>>(e0, e1, xl, pbuf, denom, out_h, E, N);

  k_h_stats<<<(N + 31) / 32, 128, 0, stream>>>(x, gat_bias, out_h, stats1, N);
  k_finalize<<<1, 128, 0, stream>>>(stats1, gamma1, beta1, ss1, 128, N);
  k_affine<<<(int)(((long long)N * 128 + 255) / 256), 256, 0, stream>>>(out_h, ss1, 128,
                                                                        (long long)N * 128);
  k_dinv<<<(N + 255) / 256, 256, 0, stream>>>(deg, N);
  {
    dim3 g((N + 127) / 128, 1);
    k_wmma_gemm<128, 32><<<g, 256, 0, stream>>>(out_h, Wg, nullptr, hW, N);
  }
  k_gcn_scatter<<<(EP + 7) / 8, 256, 0, stream>>>(e0, e1, deg, hW, zacc, E, N);
  k_z_stats<<<(N + 63) / 64, 32, 0, stream>>>(bg, zacc, stats2, N);
  k_finalize<<<1, 32, 0, stream>>>(stats2, gamma2, beta2, ss2, 32, N);
  k_affine<<<(int)(((long long)N * 32 + 255) / 256), 256, 0, stream>>>(zacc, ss2, 32,
                                                                       (long long)N * 32);
  k_decode<<<(E + 7) / 8, 256, 0, stream>>>(e0, e1, zacc, out_probs, E);
}
